// VectorQuantizer_37726992728424
// MI455X (gfx1250) — matrix operand order repaired and checked by reading, not compiled, not run on hardware
//
#include <hip/hip_runtime.h>

// ---- problem constants ----
#define KCODES 1024
#define DIM    256
#define NROWS  65536
#define ROWS_PER_BLOCK 128
#define NBLOCKS (NROWS / ROWS_PER_BLOCK)   // 512
#define LDS_STRIDE 264                     // 256 + 8 bf16 pad: row stride = 132 dwords = 4 mod 64 banks
#define QELEMS 16777216                    // 32*2048*256

typedef __attribute__((ext_vector_type(16))) __bf16 v16bf;
typedef __attribute__((ext_vector_type(8)))  __bf16 v8bf;
typedef __attribute__((ext_vector_type(4)))  __bf16 v4bf;
typedef __attribute__((ext_vector_type(8)))  float  v8f;
typedef __attribute__((ext_vector_type(4)))  float  v4f;

static __device__ __forceinline__ v16bf cat8(v8bf a, v8bf b) {
  return __builtin_shufflevector(a, b, 0,1,2,3,4,5,6,7,8,9,10,11,12,13,14,15);
}

// ---------------- Kernel 1: split W into bf16 hi/lo, compute c_k = ||w_k||^2 ----
__global__ void vq_prep(const float* __restrict__ W,
                        __bf16* __restrict__ whi, __bf16* __restrict__ wlo,
                        float* __restrict__ cvec) {
  const int wave = threadIdx.x >> 5;
  const int lane = threadIdx.x & 31;
  const int code = blockIdx.x * 8 + wave;           // grid 128 * 8 waves = 1024 codes
  const float* wr = W + code * DIM;
  float acc = 0.0f;
#pragma unroll
  for (int j = 0; j < 8; ++j) {
    const int d = j * 32 + lane;
    const float v = wr[d];
    const __bf16 h = (__bf16)v;
    const __bf16 l = (__bf16)(v - (float)h);
    whi[code * DIM + d] = h;
    wlo[code * DIM + d] = l;
    acc += v * v;
  }
#pragma unroll
  for (int off = 16; off; off >>= 1) acc += __shfl_xor(acc, off, 32);
  if (lane == 0) cvec[code] = acc;
}

// ---------------- Kernel 2: WMMA scores + argmin + gather + SSE partials ------
__global__ void __launch_bounds__(256)
vq_main(const float* __restrict__ x, const float* __restrict__ Wf,
        const __bf16* __restrict__ whi, const __bf16* __restrict__ wlo,
        const float* __restrict__ cvec,
        float* __restrict__ out_q, float* __restrict__ out_idx,
        float* __restrict__ partials) {
  __shared__ __bf16 s_hi[ROWS_PER_BLOCK * LDS_STRIDE];
  __shared__ __bf16 s_lo[ROWS_PER_BLOCK * LDS_STRIDE];
  __shared__ float  s_loss[8];

  const int tid  = threadIdx.x;
  const int lane = tid & 31;
  const int wave = tid >> 5;
  const int blockRow = (int)blockIdx.x * ROWS_PER_BLOCK;
  const float* xg = x + (size_t)blockRow * DIM;

  // ---- Stage x tile (128 x 256 f32) -> bf16 hi/lo in LDS, packed b64 stores ----
#pragma unroll
  for (int i = 0; i < 32; ++i) {
    const int off = i * 1024 + tid * 4;     // 256 thr * float4 = 1024 floats / iter
    const int row = off >> 8;
    const int col = off & 255;
    const v4f v = *(const v4f*)(xg + off);
    v4bf h, l;
#pragma unroll
    for (int e = 0; e < 4; ++e) {
      const __bf16 hb = (__bf16)v[e];
      h[e] = hb;
      l[e] = (__bf16)(v[e] - (float)hb);
    }
    const int li = row * LDS_STRIDE + col;
    *(v4bf*)(s_hi + li) = h;     // one ds_store_b64
    *(v4bf*)(s_lo + li) = l;     // one ds_store_b64
  }
  __syncthreads();

  // ---- Per-wave 16x16 C tiles over 64 code tiles; d2 = c_k - 2*x.w ----
  const int m    = lane & 15;       // A: row within tile / B,C: column (code) within tile
  const int half = lane >> 4;
  const int aRow = (wave * 16 + m) * LDS_STRIDE;

  float bestv[8];
  int   besti[8];
#pragma unroll
  for (int r = 0; r < 8; ++r) { bestv[r] = 3.4e38f; besti[r] = 0; }

  for (int kt = 0; kt < 64; ++kt) {
    // Three independent accumulator chains (one per split-bf16 term) so
    // consecutive WMMAs are not serialized on a single C RAW chain.
    v8f chh = {0.f, 0.f, 0.f, 0.f, 0.f, 0.f, 0.f, 0.f};
    v8f chl = {0.f, 0.f, 0.f, 0.f, 0.f, 0.f, 0.f, 0.f};
    v8f clh = {0.f, 0.f, 0.f, 0.f, 0.f, 0.f, 0.f, 0.f};
    const int code = kt * 16 + m;
    const __bf16* bh = whi + code * DIM + 16 * half;
    const __bf16* bl = wlo + code * DIM + 16 * half;
    // Warm the near cache with next code tile's B rows (locality=3 -> near scope).
    __builtin_prefetch(bh + 16 * DIM, 0, 3);
    __builtin_prefetch(bl + 16 * DIM, 0, 3);
#pragma unroll
    for (int ks = 0; ks < 8; ++ks) {
      const int kb = ks * 32;
      // A fragment: 16 contiguous bf16 of x row at k = kb + 16*half (matches B)
      // (loop-invariant: compiler hoists these, keeping full A hi/lo in VGPRs)
      const v8bf ah0 = *(const v8bf*)(s_hi + aRow + kb + 16 * half);
      const v8bf ah1 = *(const v8bf*)(s_hi + aRow + kb + 16 * half + 8);
      const v8bf al0 = *(const v8bf*)(s_lo + aRow + kb + 16 * half);
      const v8bf al1 = *(const v8bf*)(s_lo + aRow + kb + 16 * half + 8);
      // B fragment: 16 contiguous bf16 of code row at k = kb + 16*half
      const v8bf bh0 = *(const v8bf*)(bh + kb);
      const v8bf bh1 = *(const v8bf*)(bh + kb + 8);
      const v8bf bl0 = *(const v8bf*)(bl + kb);
      const v8bf bl1 = *(const v8bf*)(bl + kb + 8);
      const v16bf AH = cat8(ah0, ah1);
      const v16bf AL = cat8(al0, al1);
      const v16bf BH = cat8(bh0, bh1);
      const v16bf BL = cat8(bl0, bl1);
      // split-bf16: hi*hi + hi*lo + lo*hi  (~fp32-accurate dot, f32 accumulate)
      chh = __builtin_amdgcn_wmma_f32_16x16x32_bf16(false, AH, false, BH, (short)0, chh, false, false);
      chl = __builtin_amdgcn_wmma_f32_16x16x32_bf16(false, AH, false, BL, (short)0, chl, false, false);
      clh = __builtin_amdgcn_wmma_f32_16x16x32_bf16(false, AL, false, BH, (short)0, clh, false, false);
    }
    const float cn = cvec[code];
#pragma unroll
    for (int r = 0; r < 8; ++r) {
      const float s = chh[r] + chl[r] + clh[r];
      const float d2 = cn - 2.0f * s;      // row = r + 8*half
      if (d2 < bestv[r]) { bestv[r] = d2; besti[r] = code; }
    }
  }

  // ---- argmin across the 16 lanes of each half (lexicographic: first-min tie-break) ----
#pragma unroll
  for (int r = 0; r < 8; ++r) {
    float v = bestv[r];
    int   i = besti[r];
#pragma unroll
    for (int off = 8; off; off >>= 1) {
      const float ov = __shfl_xor(v, off, 32);
      const int   oi = __shfl_xor(i, off, 32);
      if (ov < v || (ov == v && oi < i)) { v = ov; i = oi; }
    }
    bestv[r] = v;
    besti[r] = i;
  }

  // ---- write idx output (as float, per harness output dtype) ----
  if (m == 0) {   // lanes 0 and 16
#pragma unroll
    for (int r = 0; r < 8; ++r) {
      const int rowg = blockRow + wave * 16 + r + 8 * half;
      out_idx[rowg] = (float)besti[r];
    }
  }

  // ---- gather W[idx] -> quantized, accumulate SSE for the loss (b128 path) ----
  float acc = 0.0f;
  const int wrow = blockRow + wave * 16;
#pragma unroll
  for (int mm = 0; mm < 16; ++mm) {
    const int kc = __shfl(besti[mm & 7], (mm & 8) ? 16 : 0, 32);
    const float* wr = Wf + (size_t)kc * DIM + lane * 8;
    const float* xr = x + (size_t)(wrow + mm) * DIM + lane * 8;
    float*       qr = out_q + (size_t)(wrow + mm) * DIM + lane * 8;
    const v4f w0 = *(const v4f*)(wr);
    const v4f w1 = *(const v4f*)(wr + 4);
    const v4f x0 = *(const v4f*)(xr);
    const v4f x1 = *(const v4f*)(xr + 4);
    *(v4f*)(qr)     = w0;
    *(v4f*)(qr + 4) = w1;
#pragma unroll
    for (int e = 0; e < 4; ++e) {
      const float d0 = w0[e] - x0[e];
      const float d1 = w1[e] - x1[e];
      acc += d0 * d0;
      acc += d1 * d1;
    }
  }
#pragma unroll
  for (int off = 16; off; off >>= 1) acc += __shfl_xor(acc, off, 32);
  if (lane == 0) s_loss[wave] = acc;
  __syncthreads();
  if (tid == 0) {
    float s = 0.0f;
#pragma unroll
    for (int w = 0; w < 8; ++w) s += s_loss[w];
    partials[blockIdx.x] = s;   // deterministic fixed-order block partial
  }
}

// ---------------- Kernel 3: deterministic final loss reduction ----------------
__global__ void vq_finalize(const float* __restrict__ partials, float* __restrict__ out_loss) {
  if (threadIdx.x == 0 && blockIdx.x == 0) {
    float s = 0.0f;
    for (int i = 0; i < NBLOCKS; ++i) s += partials[i];
    // loss = q_latent + 0.25*e_latent = 1.25 * mean((quant-x)^2)
    out_loss[0] = 1.25f * s / (float)QELEMS;
  }
}

extern "C" void kernel_launch(void* const* d_in, const int* in_sizes, int n_in,
                              void* d_out, int out_size, void* d_ws, size_t ws_size,
                              hipStream_t stream) {
  const float* x = (const float*)d_in[0];   // [32,2048,256] f32
  const float* W = (const float*)d_in[1];   // [1024,256]    f32

  char* ws = (char*)d_ws;
  __bf16* whi   = (__bf16*)(ws);                        // 512 KB
  __bf16* wlo   = (__bf16*)(ws + 524288);               // 512 KB
  float*  cvec  = (float*)(ws + 1048576);               // 4 KB
  float*  parts = (float*)(ws + 1048576 + 4096);        // 2 KB

  float* out_q    = (float*)d_out;          // [16777216]
  float* out_loss = out_q + QELEMS;         // [1]
  float* out_idx  = out_q + QELEMS + 1;     // [65536]

  vq_prep<<<KCODES / 8, 256, 0, stream>>>(W, whi, wlo, cvec);
  vq_main<<<NBLOCKS, 256, 0, stream>>>(x, W, whi, wlo, cvec, out_q, out_idx, parts);
  vq_finalize<<<1, 32, 0, stream>>>(parts, out_loss);
}
